// SelfAttention_80496277062181
// MI455X (gfx1250) — matrix operand order repaired and checked by reading, not compiled, not run on hardware
//
#include <hip/hip_runtime.h>

#define S_TOK 2048
#define DIM   1536
#define NH    12
#define HD    128
#define NB    32
#define BLK   64
#define LHS   16
#define NFR   2

typedef __attribute__((ext_vector_type(16))) _Float16 v16h;
typedef __attribute__((ext_vector_type(8)))  float    v8f;

#ifndef __has_builtin
#define __has_builtin(x) 0
#endif
#if __has_builtin(__builtin_amdgcn_tensor_load_to_lds) && \
    __has_builtin(__builtin_amdgcn_s_wait_tensorcnt)
#define USE_TDM 1
typedef unsigned int u32x4 __attribute__((ext_vector_type(4)));
typedef int          i32x8 __attribute__((ext_vector_type(8)));
typedef int          i32x4 __attribute__((ext_vector_type(4)));
#else
#define USE_TDM 0
#endif

static __device__ __forceinline__ v8f wmma_f16(v16h a, v16h b, v8f c) {
  // D = A(16x32 f16) * B(32x16 f16) + C(16x16 f32)
  return __builtin_amdgcn_wmma_f32_16x16x32_f16(false, a, false, b, (short)0, c,
                                                false, false);
}

// ---------------- f32 -> f16 convert (vector x4) ----------------
__global__ __launch_bounds__(256) void k_cvt_f16(const float* __restrict__ in,
                                                 _Float16* __restrict__ out,
                                                 int n4) {
  int i = blockIdx.x * blockDim.x + threadIdx.x;
  if (i >= n4) return;
  float4 v = reinterpret_cast<const float4*>(in)[i];
  _Float16 h[4] = {(_Float16)v.x, (_Float16)v.y, (_Float16)v.z, (_Float16)v.w};
  reinterpret_cast<float2*>(out)[i] = *reinterpret_cast<float2*>(h);
}

// ---------------- WMMA GEMM: C[MxN] = A[MxK] * B[KxN], f16 in / f32 out ------
// 64x64 tile per WG, 128 threads = 4 waves, each wave a 16x64 strip.
__global__ __launch_bounds__(128) void k_gemm_f16f32(const _Float16* __restrict__ A,
                                                     const _Float16* __restrict__ B,
                                                     float* __restrict__ C,
                                                     int M, int N, int K) {
  __shared__ _Float16 As[64][40];   // 64 x 32 (+8 pad), 80B pitch (16B aligned)
  __shared__ _Float16 Bs[32][72];   // 32 x 64 (+8 pad), 144B pitch
  const int tid  = threadIdx.x;
  const int lane = tid & 31;
  const int wave = tid >> 5;
  const int m0 = blockIdx.y * 64;
  const int n0 = blockIdx.x * 64;
  const int khalf = (lane >> 4) << 3;          // 0 or 8 (A-frag K split)
  const int arow  = wave * 16 + (lane & 15);

  v8f acc[4];
  for (int t = 0; t < 4; ++t)
    for (int e = 0; e < 8; ++e) acc[t][e] = 0.0f;

  for (int kt = 0; kt < K; kt += 32) {
    __syncthreads();
    for (int j = tid; j < 256; j += 128) {     // A tile: 64x32 halfs
      int r = j >> 2, seg = j & 3;
      *reinterpret_cast<float4*>(&As[r][seg * 8]) =
          *reinterpret_cast<const float4*>(&A[(size_t)(m0 + r) * K + kt + seg * 8]);
      if (kt + 32 < K)                         // prefetch next A tile -> global_prefetch
        __builtin_prefetch(&A[(size_t)(m0 + r) * K + kt + 32 + seg * 8], 0, 0);
    }
    for (int j = tid; j < 256; j += 128) {     // B tile: 32x64 halfs
      int r = j >> 3, seg = j & 7;
      *reinterpret_cast<float4*>(&Bs[r][seg * 8]) =
          *reinterpret_cast<const float4*>(&B[(size_t)(kt + r) * N + n0 + seg * 8]);
      if (kt + 32 < K)                         // prefetch next B tile
        __builtin_prefetch(&B[(size_t)(kt + 32 + r) * N + n0 + seg * 8], 0, 0);
    }
    __syncthreads();
    v16h a;
    for (int e = 0; e < 16; ++e) {             // A frag: lane=M, K split by half
      int kk = khalf + (e & 7) + ((e >> 3) << 4);
      a[e] = As[arow][kk];
    }
    for (int nt = 0; nt < 4; ++nt) {
      v16h b;                                  // B frag: lane=N, elems=K
      for (int e = 0; e < 16; ++e) b[e] = Bs[khalf + (e & 7) + ((e >> 3) << 4)][nt * 16 + (lane & 15)];
      acc[nt] = wmma_f16(a, b, acc[nt]);
    }
  }
  const int mw = wave * 16 + khalf;            // D: vgpr r -> row r (+8 for hi lanes)
  const int nn = lane & 15;
  for (int nt = 0; nt < 4; ++nt)
    for (int r = 0; r < 8; ++r)
      C[(size_t)(m0 + mw + r) * N + n0 + nt * 16 + nn] = acc[nt][r];
}

// ---------------- RMSNorm row + f16 convert ----------------
__global__ __launch_bounds__(256) void k_rmsnorm_f16(const float* __restrict__ in,
                                                     const float* __restrict__ w,
                                                     _Float16* __restrict__ out) {
  const int row = blockIdx.x;
  const int tid = threadIdx.x;
  __shared__ float red[256];
  float ss = 0.f;
  for (int i = tid; i < DIM; i += 256) {
    float v = in[(size_t)row * DIM + i];
    ss += v * v;
  }
  red[tid] = ss;
  __syncthreads();
  for (int s = 128; s > 0; s >>= 1) {
    if (tid < s) red[tid] += red[tid + s];
    __syncthreads();
  }
  const float sc = rsqrtf(red[0] / (float)DIM + 1e-6f);
  for (int i = tid; i < DIM; i += 256)
    out[(size_t)row * DIM + i] = (_Float16)(in[(size_t)row * DIM + i] * sc * w[i]);
}

// ---------------- 64-token average pooling ----------------
__global__ __launch_bounds__(256) void k_pool(const _Float16* __restrict__ in,
                                              float* __restrict__ out) {
  int i = blockIdx.x * blockDim.x + threadIdx.x;
  if (i >= NB * DIM) return;
  int nb = i / DIM, c = i - nb * DIM;
  float s = 0.f;
  for (int r = 0; r < BLK; ++r) s += (float)in[(size_t)(nb * BLK + r) * DIM + c];
  out[i] = s * (1.f / 64.f);
}

// ---------------- draft block mask: scores + local win + softmax + top-k -----
__global__ __launch_bounds__(256) void k_draft_mask(const float* __restrict__ qp,
                                                    const float* __restrict__ kp,
                                                    unsigned* __restrict__ mask) {
  const int g  = blockIdx.x;                   // 0..NH*NFR-1
  const int h  = g / NFR, fr = g % NFR;
  const int tid = threadIdx.x;
  __shared__ float P[LHS * NB];
  __shared__ float thr_s;
  for (int idx = tid; idx < LHS * NB; idx += 256) {
    int ql = idx >> 5, kb = idx & 31;
    int qb = fr * LHS + ql;
    const float* qv = qp + (size_t)qb * DIM + h * HD;
    const float* kv = kp + (size_t)kb * DIM + h * HD;
    float s = 0.f;
    for (int d = 0; d < HD; ++d) s += qv[d] * kv[d];
    s *= 0.08838834764831843f;                 // 1/sqrt(128)
    int qy = ql >> 2, qx = ql & 3;
    int kl = kb & 15;
    int ky = kl >> 2, kx = kl & 3;
    bool inw = (ky >= qy - 3) && (ky <= qy + 2) && (kx >= qx - 3) && (kx <= qx + 2);
    P[idx] = inw ? s : -__builtin_inff();
  }
  __syncthreads();
  if (tid < LHS) {                             // softmax over 32 key-blocks
    float mx = -__builtin_inff();
    for (int j = 0; j < NB; ++j) mx = fmaxf(mx, P[tid * NB + j]);
    float sum = 0.f;
    for (int j = 0; j < NB; ++j) {
      float e = __expf(P[tid * NB + j] - mx);
      P[tid * NB + j] = e;
      sum += e;
    }
    float inv = 1.f / sum;
    for (int j = 0; j < NB; ++j) P[tid * NB + j] *= inv;
  }
  __syncthreads();
  if (tid == 0) {                              // exact 33rd largest of 512
    float top[33];
    for (int i = 0; i < 33; ++i) top[i] = -1.f;
    for (int i = 0; i < LHS * NB; ++i) {
      float v = P[i];
      if (v > top[32]) {
        int p = 32;
        while (p > 0 && v > top[p - 1]) { top[p] = top[p - 1]; --p; }
        top[p] = v;
      }
    }
    thr_s = top[32];
  }
  __syncthreads();
  if (tid < LHS) {
    unsigned b = 0u;
    for (int j = 0; j < NB; ++j)
      if (P[tid * NB + j] > thr_s) b |= (1u << j);
    mask[h * NB + fr * LHS + tid] = b;
  }
}

// ---------------- block-sparse flash attention (WMMA) ----------------
// grid (NB, NH); 128 threads = 4 waves; wave = 16 query rows.
__global__ __launch_bounds__(128) void k_attn(const _Float16* __restrict__ Q,
                                              const _Float16* __restrict__ Kt,
                                              const _Float16* __restrict__ V,
                                              const unsigned* __restrict__ mask,
                                              float* __restrict__ O) {
  const int qb = blockIdx.x, h = blockIdx.y;
  const int tid = threadIdx.x, lane = tid & 31, wave = tid >> 5;
  __shared__ _Float16 Qs[BLK][HD];             // Q row-major
  __shared__ _Float16 KsT[HD][72];             // K TRANSPOSED: [d][n], 144B pitch
  __shared__ _Float16 Vs[BLK][HD];             // V row-major
  __shared__ _Float16 Ps[4][16][72];           // wave-private P re-layout patch
  const unsigned bits = mask[h * NB + qb];
  if (bits == 0u) {                            // row_live == false -> zeros
    for (int j = tid; j < BLK * HD; j += 128) {
      int r = j >> 7, d = j & 127;
      O[(size_t)(qb * BLK + r) * DIM + h * HD + d] = 0.f;
    }
    return;
  }
  for (int j = tid; j < 1024; j += 128) {      // Q tile 64x128 halfs
    int r = j >> 4, seg = j & 15;
    *reinterpret_cast<float4*>(&Qs[r][seg * 8]) =
        *reinterpret_cast<const float4*>(&Q[(size_t)(qb * BLK + r) * DIM + h * HD + seg * 8]);
  }
  __syncthreads();
  const int khalf = (lane >> 4) << 3;
  const int rowA  = wave * 16 + (lane & 15);
  v16h qf[4];                                  // Q frags, K-dim chunks of 32
  for (int kc = 0; kc < 4; ++kc)
    for (int e = 0; e < 16; ++e) {
      int kk = kc * 32 + khalf + (e & 7) + ((e >> 3) << 4);
      qf[kc][e] = Qs[rowA][kk];
    }
  v8f oacc[8];
  for (int t = 0; t < 8; ++t)
    for (int e = 0; e < 8; ++e) oacc[t][e] = 0.f;
  float rmax[8], rsum[8];
  for (int r = 0; r < 8; ++r) { rmax[r] = -__builtin_inff(); rsum[r] = 0.f; }
  const float sc = 0.08838834764831843f;

  for (int kb = 0; kb < NB; ++kb) {
    if (!((bits >> kb) & 1u)) continue;        // uniform across block
    __syncthreads();
#if USE_TDM
    // ---- V tile via Tensor Data Mover: 2D tile 128x64 of 2B elems ----
    if (wave == 0) {
      unsigned long long ga =
          (unsigned long long)(uintptr_t)&V[(size_t)(kb * BLK) * DIM + h * HD];
      unsigned ldsa = (unsigned)(uintptr_t)&Vs[0][0];
      u32x4 g0;
      g0[0] = 1u;                                           // count=1 (valid D#)
      g0[1] = ldsa;                                         // lds_addr
      g0[2] = (unsigned)(ga & 0xffffffffu);                 // global_addr[31:0]
      g0[3] = (unsigned)((ga >> 32) & 0x1ffffffu) | (2u << 30); // [56:32]|type=2
      i32x8 g1;
      g1[0] = (1 << 16);                                    // data_size=2B, wg_mask=0
      g1[1] = (int)((unsigned)HD << 16);                    // tensor_dim0 = 128
      g1[2] = (int)((unsigned)BLK << 16);                   // tensor_dim1 = 64
      g1[3] = (int)((unsigned)HD << 16);                    // tile_dim0 = 128
      g1[4] = BLK;                                          // tile_dim1=64, tile_dim2=0
      g1[5] = DIM;                                          // tensor_dim0_stride = 1536
      g1[6] = 0;
      g1[7] = 0;
      i32x4 gz; gz[0] = 0; gz[1] = 0; gz[2] = 0; gz[3] = 0;
      i32x8 gz8;
      for (int t = 0; t < 8; ++t) gz8[t] = 0;
      __builtin_amdgcn_tensor_load_to_lds(g0, g1, gz, gz, gz8, 0);
      __builtin_amdgcn_s_wait_tensorcnt(0);
    }
#else
    for (int j = tid; j < 1024; j += 128) {    // V tile 64x128 halfs
      int r = j >> 4, seg = j & 15;
      *reinterpret_cast<float4*>(&Vs[r][seg * 8]) =
          *reinterpret_cast<const float4*>(&V[(size_t)(kb * BLK + r) * DIM + h * HD + seg * 8]);
    }
#endif
    for (int j = tid; j < 1024; j += 128) {    // K tile, stored TRANSPOSED
      int r = j >> 4, seg = j & 15;            // r = key index, seg*8 = dim offset
      float4 kv4 = *reinterpret_cast<const float4*>(
          &Kt[(size_t)(kb * BLK + r) * DIM + h * HD + seg * 8]);
      _Float16 hk[8];
      *reinterpret_cast<float4*>(hk) = kv4;
      for (int t = 0; t < 8; ++t) KsT[seg * 8 + t][r] = hk[t];
    }
    __syncthreads();
    v8f sacc[4];                               // S strip 16x64
    for (int t = 0; t < 4; ++t)
      for (int e = 0; e < 8; ++e) sacc[t][e] = 0.f;
    for (int nt = 0; nt < 4; ++nt)
      for (int kc = 0; kc < 4; ++kc) {
        v16h b;                                // B[d][n]=K[n][d]: contiguous in KsT
        for (int e = 0; e < 16; ++e) b[e] = KsT[kc * 32 + khalf + (e & 7) + ((e >> 3) << 4)][nt * 16 + (lane & 15)];
        sacc[nt] = wmma_f16(qf[kc], b, sacc[nt]);
      }
    for (int r = 0; r < 8; ++r) {              // online softmax per row
      float mx = -__builtin_inff();
      for (int nt = 0; nt < 4; ++nt) {
        sacc[nt][r] *= sc;
        mx = fmaxf(mx, sacc[nt][r]);
      }
      for (int off = 1; off < 16; off <<= 1) mx = fmaxf(mx, __shfl_xor(mx, off, 32));
      float nm = fmaxf(rmax[r], mx);
      float corr = __expf(rmax[r] - nm);
      rmax[r] = nm;
      float rs = 0.f;
      for (int nt = 0; nt < 4; ++nt) {
        float e = __expf(sacc[nt][r] - nm);
        sacc[nt][r] = e;
        rs += e;
      }
      for (int off = 1; off < 16; off <<= 1) rs += __shfl_xor(rs, off, 32);
      rsum[r] = rsum[r] * corr + rs;
      for (int t = 0; t < 8; ++t) oacc[t][r] *= corr;
    }
    // P (D-layout) -> LDS -> A-layout frags; wave-private, in-order DS
    for (int nt = 0; nt < 4; ++nt)
      for (int r = 0; r < 8; ++r)
        Ps[wave][r + khalf][nt * 16 + (lane & 15)] = (_Float16)sacc[nt][r];
    v16h pf[2];
    for (int kc = 0; kc < 2; ++kc)
      for (int e = 0; e < 16; ++e) {
        int kk = kc * 32 + khalf + (e & 7) + ((e >> 3) << 4);
        pf[kc][e] = Ps[wave][lane & 15][kk];
      }
    for (int dt = 0; dt < 8; ++dt)             // O += P(16x64) * V(64x128)
      for (int kc = 0; kc < 2; ++kc) {
        v16h b;                                // B[k][n]=V[k][n]: lane=n, elem=k
        for (int e = 0; e < 16; ++e) b[e] = Vs[kc * 32 + khalf + (e & 7) + ((e >> 3) << 4)][dt * 16 + (lane & 15)];
        oacc[dt] = wmma_f16(pf[kc], b, oacc[dt]);
      }
  }
  for (int dt = 0; dt < 8; ++dt)
    for (int r = 0; r < 8; ++r) {
      int m = wave * 16 + khalf + r;
      int d = dt * 16 + (lane & 15);
      O[(size_t)(qb * BLK + m) * DIM + h * HD + d] = oacc[dt][r] / rsum[r];
    }
}

extern "C" void kernel_launch(void* const* d_in, const int* in_sizes, int n_in,
                              void* d_out, int out_size, void* d_ws, size_t ws_size,
                              hipStream_t stream) {
  (void)in_sizes; (void)n_in; (void)out_size; (void)ws_size;
  const float* x   = (const float*)d_in[0];
  const float* wq  = (const float*)d_in[1];
  const float* wk  = (const float*)d_in[2];
  const float* wv  = (const float*)d_in[3];
  const float* wo  = (const float*)d_in[4];
  const float* qnw = (const float*)d_in[5];
  const float* knw = (const float*)d_in[6];
  float* out = (float*)d_out;

  char* ws = (char*)d_ws;
  size_t off = 0;
  auto alloc = [&](size_t bytes) -> void* {
    void* p = ws + off;
    off += (bytes + 255) & ~(size_t)255;
    return p;
  };
  const size_t SD = (size_t)S_TOK * DIM;
  const size_t DD = (size_t)DIM * DIM;
  _Float16* xh  = (_Float16*)alloc(SD * 2);
  _Float16* wqh = (_Float16*)alloc(DD * 2);
  _Float16* wkh = (_Float16*)alloc(DD * 2);
  _Float16* wvh = (_Float16*)alloc(DD * 2);
  _Float16* woh = (_Float16*)alloc(DD * 2);
  float*    qf  = (float*)alloc(SD * 4);
  float*    kf  = (float*)alloc(SD * 4);
  float*    vf  = (float*)alloc(SD * 4);
  _Float16* qh  = (_Float16*)alloc(SD * 2);
  _Float16* kh  = (_Float16*)alloc(SD * 2);
  _Float16* vh  = (_Float16*)alloc(SD * 2);
  float*    of  = (float*)alloc(SD * 4);
  _Float16* oh  = (_Float16*)alloc(SD * 2);
  float*    qp  = (float*)alloc((size_t)NB * DIM * 4);
  float*    kp  = (float*)alloc((size_t)NB * DIM * 4);
  unsigned* msk = (unsigned*)alloc((size_t)NH * NB * 4);

  // 1) f16 conversions of x and weights
  k_cvt_f16<<<(int)(SD / 4 / 256), 256, 0, stream>>>(x,  xh,  (int)(SD / 4));
  k_cvt_f16<<<(int)(DD / 4 / 256), 256, 0, stream>>>(wq, wqh, (int)(DD / 4));
  k_cvt_f16<<<(int)(DD / 4 / 256), 256, 0, stream>>>(wk, wkh, (int)(DD / 4));
  k_cvt_f16<<<(int)(DD / 4 / 256), 256, 0, stream>>>(wv, wvh, (int)(DD / 4));
  k_cvt_f16<<<(int)(DD / 4 / 256), 256, 0, stream>>>(wo, woh, (int)(DD / 4));

  // 2) Q/K/V projections (WMMA)
  dim3 gg(DIM / 64, S_TOK / 64);
  k_gemm_f16f32<<<gg, 128, 0, stream>>>(xh, wqh, qf, S_TOK, DIM, DIM);
  k_gemm_f16f32<<<gg, 128, 0, stream>>>(xh, wkh, kf, S_TOK, DIM, DIM);
  k_gemm_f16f32<<<gg, 128, 0, stream>>>(xh, wvh, vf, S_TOK, DIM, DIM);

  // 3) RMSNorm(q/k) -> f16; convert v -> f16
  k_rmsnorm_f16<<<S_TOK, 256, 0, stream>>>(qf, qnw, qh);
  k_rmsnorm_f16<<<S_TOK, 256, 0, stream>>>(kf, knw, kh);
  k_cvt_f16<<<(int)(SD / 4 / 256), 256, 0, stream>>>(vf, vh, (int)(SD / 4));

  // 4) 64-token pooling
  k_pool<<<(NB * DIM + 255) / 256, 256, 0, stream>>>(qh, qp);
  k_pool<<<(NB * DIM + 255) / 256, 256, 0, stream>>>(kh, kp);

  // 5) draft block-sparse mask
  k_draft_mask<<<NH * NFR, 256, 0, stream>>>(qp, kp, msk);

  // 6) block-sparse flash attention
  k_attn<<<dim3(NB, NH), 128, 0, stream>>>(qh, kh, vh, msk, of);

  // 7) output projection
  k_cvt_f16<<<(int)(SD / 4 / 256), 256, 0, stream>>>(of, oh, (int)(SD / 4));
  k_gemm_f16f32<<<gg, 128, 0, stream>>>(oh, woh, out, S_TOK, DIM, DIM);
}
